// GLATE_76252849373291
// MI455X (gfx1250) — matrix operand order repaired and checked by reading, not compiled, not run on hardware
//
#include <hip/hip_runtime.h>

// ---------------------------------------------------------------------------
// Types for CDNA5 WMMA
// ---------------------------------------------------------------------------
typedef __attribute__((ext_vector_type(16))) __bf16 v16bf;
typedef __attribute__((ext_vector_type(8)))  float  v8f;
typedef __attribute__((ext_vector_type(2)))  __bf16 bf16x2;
typedef __attribute__((ext_vector_type(4)))  int    v4i;

#define AS1 __attribute__((address_space(1)))
#define AS3 __attribute__((address_space(3)))

union AFrag {
    v16bf v;
    unsigned int u[8];
};

struct alignas(16) U4 { unsigned int x, y, z, w; };

// ---------------------------------------------------------------------------
// gfx1250 async load-to-LDS (ASYNCcnt path) with safe fallback
// ---------------------------------------------------------------------------
#if defined(__AMDGCN__) && __has_builtin(__builtin_amdgcn_global_load_async_to_lds_b128)
#define HAVE_ASYNC_LDS 1
#else
#define HAVE_ASYNC_LDS 0
#endif

__device__ __forceinline__ void async_copy16(const void* gsrc, void* lds_dst) {
#if HAVE_ASYNC_LDS
    __builtin_amdgcn_global_load_async_to_lds_b128(
        (AS1 v4i*)gsrc, (AS3 v4i*)lds_dst, 0, 0);
#else
    *(U4*)lds_dst = *(const U4*)gsrc;      // global_load_b128 + ds_store_b128
#endif
}

__device__ __forceinline__ void async_wait() {
#if HAVE_ASYNC_LDS
#if __has_builtin(__builtin_amdgcn_s_wait_asynccnt)
    __builtin_amdgcn_s_wait_asynccnt(0);
#else
    asm volatile("s_wait_asynccnt 0x0" ::: "memory");
#endif
#endif
}

// ---------------------------------------------------------------------------
// Utility kernels
// ---------------------------------------------------------------------------
__global__ void zero_f32(float* __restrict__ p, long n) {
    long i = (long)blockIdx.x * blockDim.x + threadIdx.x;
    if (i < n) p[i] = 0.0f;
}

__global__ void f32_to_bf16(const float* __restrict__ in, __bf16* __restrict__ out, long n) {
    long i = (long)blockIdx.x * blockDim.x + threadIdx.x;
    if (i < n) out[i] = (__bf16)in[i];
}

// Transpose + convert weight [K, OUT] (row-major, y = x @ W) -> bf16 [OUT, K]
__global__ void wt_to_bf16_t(const float* __restrict__ w, __bf16* __restrict__ wt,
                             int K, int OUT) {
    long i = (long)blockIdx.x * blockDim.x + threadIdx.x;
    if (i >= (long)K * OUT) return;
    int k = (int)(i / OUT);
    int n = (int)(i % OUT);
    wt[(long)n * K + k] = (__bf16)w[i];
}

__global__ void deg_kernel(const int* __restrict__ dst, float* __restrict__ deg, int E) {
    int e = blockIdx.x * blockDim.x + threadIdx.x;
    if (e < E) atomicAdd(&deg[dst[e]], 1.0f);
}

// Scatter-add mean-aggregation numerator: agg[dst] += x[src]
// bf16 gather (dword-vectorized), fp32 atomic accumulation (L2-resident).
__global__ void agg_kernel(const __bf16* __restrict__ xb, const int* __restrict__ src,
                           const int* __restrict__ dst, float* __restrict__ agg, int D) {
    int e = blockIdx.x;
    int s = src[e];
    int d = dst[e];
    const bf16x2* xr = (const bf16x2*)(xb + (long)s * D);
    float* ar = agg + (long)d * D;
    int D2 = D >> 1;
    for (int f = threadIdx.x; f < D2; f += blockDim.x) {
        bf16x2 v = xr[f];
        atomicAdd(&ar[2 * f + 0], (float)v.x);
        atomicAdd(&ar[2 * f + 1], (float)v.y);
    }
}

// mean = agg / max(deg,1)  -> bf16
__global__ void mean_kernel(const float* __restrict__ agg, const float* __restrict__ deg,
                            __bf16* __restrict__ mb, long n, int D) {
    long i = (long)blockIdx.x * blockDim.x + threadIdx.x;
    if (i >= n) return;
    float c = deg[i / D];
    c = (c < 1.0f) ? 1.0f : c;
    mb[i] = (__bf16)(agg[i] / c);
}

// cur = a + b ; curb = bf16(cur)
__global__ void add2_kernel(const float* __restrict__ a, const float* __restrict__ b,
                            float* __restrict__ cur, __bf16* __restrict__ curb, long n) {
    long i = (long)blockIdx.x * blockDim.x + threadIdx.x;
    if (i >= n) return;
    float v = a[i] + b[i];
    cur[i] = v;
    curb[i] = (__bf16)v;
}

// cur = a + b + c ; curb = bf16(cur)
__global__ void add3_kernel(const float* __restrict__ a, const float* __restrict__ b,
                            const float* __restrict__ c, float* __restrict__ cur,
                            __bf16* __restrict__ curb, long n) {
    long i = (long)blockIdx.x * blockDim.x + threadIdx.x;
    if (i >= n) return;
    float v = a[i] + b[i] + c[i];
    cur[i] = v;
    curb[i] = (__bf16)v;
}

// ---------------------------------------------------------------------------
// WMMA GEMM: C[N,OUT] (+)= A[N,K] @ W[K,OUT], W given bf16-transposed [OUT,K].
// Block = 256 threads = 8 waves; block tile = 128 rows x 64 cols; each wave
// owns a 16x64 strip. The 64x32 weight slab per k-step is shared by all 8
// waves, staged in LDS (double-buffered, async-to-LDS when available).
// LDS row stride = 40 bf16 (80 B): 20-bank stride -> conflict-free reads,
// 16 B aligned for ds_load_b128.
// ---------------------------------------------------------------------------
#define BROW_STRIDE 40

__global__ __launch_bounds__(256)
void gemm_bf16_wmma(const __bf16* __restrict__ A, const __bf16* __restrict__ Wt,
                    const float* __restrict__ bias, const float* __restrict__ aptr,
                    float* __restrict__ C, int Nrows, int K, int OUT,
                    int accumulate, int prelu) {
    __shared__ __align__(16) __bf16 Bs[2][64 * BROW_STRIDE];

    const int wave = threadIdx.x >> 5;
    const int lane = threadIdx.x & 31;
    const int row0 = blockIdx.x * 128 + wave * 16;
    const int col0 = blockIdx.y * 64;

    // Keep every wave alive (barriers + WMMA need uniform participation);
    // clamp loads for out-of-range waves, skip their stores.
    const bool active = (row0 < Nrows);
    const int row0c = active ? row0 : (Nrows - 16);

    const int m    = lane & 15;   // row within A tile / col within B tile
    const int half = lane >> 4;

    const __bf16* Arow = A + (long)(row0c + m) * K;

    // Cooperative B-slab staging: thread t loads 16 B:
    //   slab row (column of W) = t>>2, 8-bf16 segment = t&3
    const int brow = threadIdx.x >> 2;
    const int bseg = (threadIdx.x & 3) * 8;
    const __bf16* bsrc_base = Wt + (long)(col0 + brow) * K + bseg;

    v8f acc[4];
    #pragma unroll
    for (int j = 0; j < 4; ++j)
        #pragma unroll
        for (int r = 0; r < 8; ++r) acc[j][r] = 0.0f;

    const int steps = K >> 5;  // K/32

    // Prime buffer 0
    async_copy16(bsrc_base, &Bs[0][brow * BROW_STRIDE + bseg]);

    for (int kt = 0; kt < steps; ++kt) {
        const int cur = kt & 1;
        const int k0  = kt << 5;

        async_wait();
        __syncthreads();                 // slab[cur] ready for all waves

        if (kt + 1 < steps) {            // prefetch next slab into other buffer
            async_copy16(bsrc_base + (k0 + 32),
                         &Bs[cur ^ 1][brow * BROW_STRIDE + bseg]);
        }

        // A fragment: 16x32 bf16 from global (the HBM stream).
        //   p<4 : K = p*2 + half*8 ; p>=4: K = 16 + (p-4)*2 + half*8
        AFrag a;
        #pragma unroll
        for (int p = 0; p < 8; ++p) {
            int kk = ((p < 4) ? 0 : 16) + (p & 3) * 2 + half * 8;
            a.u[p] = *(const unsigned int*)(Arow + k0 + kk);
        }

        #pragma unroll
        for (int j = 0; j < 4; ++j) {
            // B fragment from LDS: same K order as A fragment.
            const __bf16* bp = &Bs[cur][(j * 16 + m) * BROW_STRIDE];
            AFrag b;
            #pragma unroll
            for (int p = 0; p < 8; ++p)
                b.u[p] = *(const unsigned int*)(bp + ((p < 4) ? 0 : 16) + (p & 3) * 2 + half * 8);
            acc[j] = __builtin_amdgcn_wmma_f32_16x16x32_bf16(
                false, a.v, false, b.v, (short)0, acc[j], false, false);
        }
    }

    if (!active) return;   // wave-uniform; all barriers already passed

    const float slope = prelu ? *aptr : 0.0f;

    #pragma unroll
    for (int j = 0; j < 4; ++j) {
        const int n = col0 + j * 16 + m;
        const float bv = bias ? bias[n] : 0.0f;
        #pragma unroll
        for (int r = 0; r < 8; ++r) {
            const int row = row0 + r + half * 8;   // C/D layout: VGPR r -> M=r (+8 hi half)
            const long idx = (long)row * OUT + n;
            float v = acc[j][r] + bv;
            if (accumulate) v += C[idx];
            if (prelu) v = (v >= 0.0f) ? v : slope * v;
            C[idx] = v;
        }
    }
}

// ---------------------------------------------------------------------------
// Host-side launch
// ---------------------------------------------------------------------------
static inline long cdiv(long a, long b) { return (a + b - 1) / b; }

extern "C" void kernel_launch(void* const* d_in, const int* in_sizes, int n_in,
                              void* d_out, int out_size, void* d_ws, size_t ws_size,
                              hipStream_t stream) {
    const int DIN = 128, D = 512;
    const int N = in_sizes[0] / DIN;
    const int E = in_sizes[2] / 2;

    const float* x1   = (const float*)d_in[0];
    const float* x2   = (const float*)d_in[1];
    const int*   ei1  = (const int*)d_in[2];
    const int*   ei2  = (const int*)d_in[3];
    const float* l1Wl = (const float*)d_in[4];
    const float* l1bl = (const float*)d_in[5];
    const float* l1Wr = (const float*)d_in[6];
    const float* l2Wl = (const float*)d_in[7];
    const float* l2bl = (const float*)d_in[8];
    const float* l2Wr = (const float*)d_in[9];
    const float* l3Wl = (const float*)d_in[10];
    const float* l3bl = (const float*)d_in[11];
    const float* l3Wr = (const float*)d_in[12];
    const float* Ww   = (const float*)d_in[13];
    const float* Wb   = (const float*)d_in[14];
    const float* W2w  = (const float*)d_in[15];
    const float* W2b  = (const float*)d_in[16];
    const float* pa   = (const float*)d_in[17];

    float* out = (float*)d_out;

    // ---- carve workspace -------------------------------------------------
    char* wp = (char*)d_ws;
    auto carve = [&](size_t bytes) -> void* {
        void* r = (void*)wp;
        wp += (bytes + 255) & ~(size_t)255;
        return r;
    };
    __bf16* wl1t = (__bf16*)carve((size_t)DIN * D * 2);
    __bf16* wr1t = (__bf16*)carve((size_t)DIN * D * 2);
    __bf16* wwt  = (__bf16*)carve((size_t)DIN * D * 2);
    __bf16* w2wt = (__bf16*)carve((size_t)DIN * D * 2);
    __bf16* wl2t = (__bf16*)carve((size_t)D * D * 2);
    __bf16* wr2t = (__bf16*)carve((size_t)D * D * 2);
    __bf16* wl3t = (__bf16*)carve((size_t)D * D * 2);
    __bf16* wr3t = (__bf16*)carve((size_t)D * D * 2);
    __bf16* xb    = (__bf16*)carve((size_t)N * DIN * 2);
    float*  deg   = (float*)carve((size_t)N * 4);
    float*  agg   = (float*)carve((size_t)N * D * 4);   // also reused as GEMM temp 't'
    __bf16* meanb = (__bf16*)carve((size_t)N * D * 2);
    float*  h1    = (float*)carve((size_t)N * D * 4);
    float*  h2    = (float*)carve((size_t)N * D * 4);
    float*  cur   = (float*)carve((size_t)N * D * 4);
    __bf16* curb  = (__bf16*)carve((size_t)N * D * 2);
    float*  t     = agg;  // alias: free between mean extraction and next zeroing

    const long ND   = (long)N * D;
    const long NDin = (long)N * DIN;

    // ---- one-time weight conversions (bf16, transposed [OUT,K]) ---------
    {
        long nw = (long)DIN * D;
        wt_to_bf16_t<<<cdiv(nw, 256), 256, 0, stream>>>(l1Wl, wl1t, DIN, D);
        wt_to_bf16_t<<<cdiv(nw, 256), 256, 0, stream>>>(l1Wr, wr1t, DIN, D);
        wt_to_bf16_t<<<cdiv(nw, 256), 256, 0, stream>>>(Ww,   wwt,  DIN, D);
        wt_to_bf16_t<<<cdiv(nw, 256), 256, 0, stream>>>(W2w,  w2wt, DIN, D);
        long nw2 = (long)D * D;
        wt_to_bf16_t<<<cdiv(nw2, 256), 256, 0, stream>>>(l2Wl, wl2t, D, D);
        wt_to_bf16_t<<<cdiv(nw2, 256), 256, 0, stream>>>(l2Wr, wr2t, D, D);
        wt_to_bf16_t<<<cdiv(nw2, 256), 256, 0, stream>>>(l3Wl, wl3t, D, D);
        wt_to_bf16_t<<<cdiv(nw2, 256), 256, 0, stream>>>(l3Wr, wr3t, D, D);
    }

    dim3 gemmBlock(256);
    dim3 gemmGrid((unsigned)cdiv(N, 128), (unsigned)(D / 64));

    auto encode = [&](const float* x, const int* ei, float* z) {
        const int* src = ei;       // edge_index[0]
        const int* dst = ei + E;   // edge_index[1]

        // x -> bf16
        f32_to_bf16<<<cdiv(NDin, 256), 256, 0, stream>>>(x, xb, NDin);

        // degrees
        zero_f32<<<cdiv((long)N, 256), 256, 0, stream>>>(deg, N);
        deg_kernel<<<cdiv((long)E, 256), 256, 0, stream>>>(dst, deg, E);

        // ---------------- layer 1 (DIN -> D) ----------------
        zero_f32<<<cdiv(NDin, 256), 256, 0, stream>>>(agg, NDin);
        agg_kernel<<<E, 64, 0, stream>>>(xb, src, dst, agg, DIN);
        mean_kernel<<<cdiv(NDin, 256), 256, 0, stream>>>(agg, deg, meanb, NDin, DIN);
        // h1 = mean @ Wl1 + bl1
        gemm_bf16_wmma<<<gemmGrid, gemmBlock, 0, stream>>>(meanb, wl1t, l1bl, pa, h1,
                                                           N, DIN, D, 0, 0);
        // h1 += x @ Wr1 ; prelu
        gemm_bf16_wmma<<<gemmGrid, gemmBlock, 0, stream>>>(xb, wr1t, nullptr, pa, h1,
                                                           N, DIN, D, 1, 1);

        // in2 = h1 + x @ W_w + W_b
        gemm_bf16_wmma<<<gemmGrid, gemmBlock, 0, stream>>>(xb, wwt, Wb, pa, t,
                                                           N, DIN, D, 0, 0);
        add2_kernel<<<cdiv(ND, 256), 256, 0, stream>>>(h1, t, cur, curb, ND);

        // ---------------- layer 2 (D -> D) ------------------
        zero_f32<<<cdiv(ND, 256), 256, 0, stream>>>(agg, ND);
        agg_kernel<<<E, 256, 0, stream>>>(curb, src, dst, agg, D);
        mean_kernel<<<cdiv(ND, 256), 256, 0, stream>>>(agg, deg, meanb, ND, D);
        gemm_bf16_wmma<<<gemmGrid, gemmBlock, 0, stream>>>(meanb, wl2t, l2bl, pa, h2,
                                                           N, D, D, 0, 0);
        gemm_bf16_wmma<<<gemmGrid, gemmBlock, 0, stream>>>(curb, wr2t, nullptr, pa, h2,
                                                           N, D, D, 1, 1);

        // in3 = h2 + h1 + x @ W2_w + W2_b
        gemm_bf16_wmma<<<gemmGrid, gemmBlock, 0, stream>>>(xb, w2wt, W2b, pa, t,
                                                           N, DIN, D, 0, 0);
        add3_kernel<<<cdiv(ND, 256), 256, 0, stream>>>(h2, h1, t, cur, curb, ND);

        // ---------------- layer 3 (D -> D) ------------------
        zero_f32<<<cdiv(ND, 256), 256, 0, stream>>>(agg, ND);
        agg_kernel<<<E, 256, 0, stream>>>(curb, src, dst, agg, D);
        mean_kernel<<<cdiv(ND, 256), 256, 0, stream>>>(agg, deg, meanb, ND, D);
        gemm_bf16_wmma<<<gemmGrid, gemmBlock, 0, stream>>>(meanb, wl3t, l3bl, pa, z,
                                                           N, D, D, 0, 0);
        gemm_bf16_wmma<<<gemmGrid, gemmBlock, 0, stream>>>(curb, wr3t, nullptr, pa, z,
                                                           N, D, D, 1, 1);
    };

    encode(x1, ei1, out);            // z1
    encode(x2, ei2, out + ND);       // z2
}
